// WideResNet_ours_l5_cejsd_wtconst_4664334483971
// MI455X (gfx1250) — compile-verified, matrix operand order checked by reading, not run on hardware
//
#include <hip/hip_runtime.h>
#include <hip/hip_bf16.h>
#include <math.h>

typedef __attribute__((ext_vector_type(16))) _Float16 v16h;
typedef __attribute__((ext_vector_type(8)))  _Float16 v8h;
typedef __attribute__((ext_vector_type(8)))  float    v8f;

#define FEAT  512
#define BATCH 8192

// ---------------- wave helpers (wave32) ----------------
__device__ __forceinline__ float waveSum(float v) {
#pragma unroll
  for (int o = 16; o > 0; o >>= 1) v += __shfl_xor(v, o, 32);
  return v;
}
__device__ __forceinline__ float waveMax(float v) {
#pragma unroll
  for (int o = 16; o > 0; o >>= 1) v = fmaxf(v, __shfl_xor(v, o, 32));
  return v;
}

__device__ __forceinline__ v8f wmma_f16(v16h a, v16h b, v8f c) {
  // (neg_a, A, neg_b, B, c_mod, C, reuse_a, reuse_b)
  return __builtin_amdgcn_wmma_f32_16x16x32_f16(false, a, false, b, (short)0, c,
                                                false, false);
}

__device__ __forceinline__ v16h load_frag(const _Float16* base, int off0, int off1) {
  v8h lo = *(const v8h*)(base + off0);
  v8h hi = *(const v8h*)(base + off1);
  return __builtin_shufflevector(lo, hi, 0, 1, 2, 3, 4, 5, 6, 7,
                                 8, 9, 10, 11, 12, 13, 14, 15);
}

// ---------------- misc small kernels ----------------
__global__ void zero_accum_kernel(float* acc) {
  if (threadIdx.x < 16) acc[threadIdx.x] = 0.0f;
}

__global__ void finalize_kernel(const float* acc, float* out_loss) {
  // acc[0]=ce, acc[1..4]=jsd, acc[5..8]=cosine
  *out_loss = acc[0] + (acc[1] + acc[2] + acc[3] + acc[4])
                     - (acc[5] + acc[6] + acc[7] + acc[8]);
}

// ---------------- weight row normalization (one wave per row) ----------------
__global__ __launch_bounds__(32)
void norm_rows_kernel(const float* __restrict__ W, float* __restrict__ wn,
                      _Float16* __restrict__ wh, int C) {
  const int row  = blockIdx.x;
  const int lane = threadIdx.x;
  size_t base = (size_t)row * FEAT;
  if (row >= C) {  // zero pad rows so GEMM tiles read zeros
#pragma unroll
    for (int t = 0; t < 16; t++) {
      int f = lane + 32 * t;
      wn[base + f] = 0.0f;
      wh[base + f] = (_Float16)0.0f;
    }
    return;
  }
  float vals[16];
  float s = 0.0f;
#pragma unroll
  for (int t = 0; t < 16; t++) {
    float v = W[base + lane + 32 * t];
    vals[t] = v;
    s += v * v;
  }
  s = waveSum(s);
  float inv = 1.0f / fmaxf(sqrtf(s), 1e-12f);
#pragma unroll
  for (int t = 0; t < 16; t++) {
    float v = vals[t] * inv;
    wn[base + lane + 32 * t] = v;
    wh[base + lane + 32 * t] = (_Float16)v;
  }
}

// ---------------- x fp32 -> f16 ----------------
__global__ __launch_bounds__(256)
void convert_x_kernel(const float* __restrict__ x, _Float16* __restrict__ xh) {
  size_t i = (size_t)blockIdx.x * 256 + threadIdx.x;
  if (i < (size_t)BATCH * FEAT) xh[i] = (_Float16)x[i];
}

// ---------------- WMMA GEMM: Out[8192,C] = xh @ whT + bias ----------------
// block = 256 threads (8 waves); block tile 128(M) x 128(N); wave tile 32(M) x 64(N)
// Each wave: 2 A-fragments x 4 B-fragments -> 8 WMMAs per K-step (6 b128-pair loads).
__global__ __launch_bounds__(256)
void gemm_wmma_kernel(const _Float16* __restrict__ A,   // [8192,512]
                      const _Float16* __restrict__ Bw,  // [Cpad,512] (row = class)
                      const float* __restrict__ bias,   // [C]
                      float* __restrict__ Out,          // [8192,C]
                      int C) {
  const int lane = threadIdx.x & 31;
  const int wave = threadIdx.x >> 5;
  const int wm = wave & 3;   // 0..3 -> M strip (32 rows each)
  const int wn = wave >> 2;  // 0..1 -> N strip (64 cols each)
  const int m0 = blockIdx.y * 128 + wm * 32;
  const int n0 = blockIdx.x * 128 + wn * 64;
  const int r = lane & 15;
  const int h = lane >> 4;

  v8f acc[2][4];
#pragma unroll
  for (int i = 0; i < 2; i++)
#pragma unroll
    for (int j = 0; j < 4; j++) acc[i][j] = (v8f){};

  const _Float16* arow[2];
  arow[0] = A + (size_t)(m0 + r)      * FEAT;
  arow[1] = A + (size_t)(m0 + 16 + r) * FEAT;
  const _Float16* brow[4];
#pragma unroll
  for (int j = 0; j < 4; j++) brow[j] = Bw + (size_t)(n0 + j * 16 + r) * FEAT;

  for (int k0 = 0; k0 < FEAT; k0 += 32) {
    const int off0 = k0 + 8 * h;       // K = k0+8h .. +7
    const int off1 = k0 + 16 + 8 * h;  // K = k0+16+8h .. +7
    v16h af[2], bf[4];
#pragma unroll
    for (int i = 0; i < 2; i++) af[i] = load_frag(arow[i], off0, off1);
#pragma unroll
    for (int j = 0; j < 4; j++) bf[j] = load_frag(brow[j], off0, off1);
#pragma unroll
    for (int i = 0; i < 2; i++)
#pragma unroll
      for (int j = 0; j < 4; j++) acc[i][j] = wmma_f16(af[i], bf[j], acc[i][j]);
  }

  // D layout: lane -> col = n0 + j*16 + (lane&15); VGPR v -> row = m0 + i*16 + v + 8*(lane>>4)
#pragma unroll
  for (int i = 0; i < 2; i++) {
#pragma unroll
    for (int j = 0; j < 4; j++) {
      int col = n0 + j * 16 + r;
      if (col < C) {
        float bvv = bias[col];
#pragma unroll
        for (int v = 0; v < 8; v++) {
          int rowi = m0 + i * 16 + v + 8 * h;
          Out[(size_t)rowi * C + col] = acc[i][j][v] + bvv;
        }
      }
    }
  }
}

// ---------------- in-place softmax per row; optional CE accumulation ----------------
__global__ __launch_bounds__(256)
void softmax_kernel(float* __restrict__ logits, int C,
                    const int* __restrict__ targets, float* __restrict__ accum,
                    float invB) {
  const int row = blockIdx.x;
  float* rp = logits + (size_t)row * C;
  const int tid = threadIdx.x, lane = tid & 31, wv = tid >> 5;
  __shared__ float sred[8];
  __shared__ float sMax, sSum;

  float mx = -3.4e38f;
  for (int c = tid; c < C; c += 256) mx = fmaxf(mx, rp[c]);
  mx = waveMax(mx);
  if (lane == 0) sred[wv] = mx;
  __syncthreads();
  if (tid == 0) {
    float m = sred[0];
    for (int i = 1; i < 8; i++) m = fmaxf(m, sred[i]);
    sMax = m;
  }
  __syncthreads();
  mx = sMax;
  __syncthreads();

  float s = 0.0f;
  for (int c = tid; c < C; c += 256) s += expf(rp[c] - mx);
  s = waveSum(s);
  if (lane == 0) sred[wv] = s;
  __syncthreads();
  if (tid == 0) {
    float t = 0.0f;
    for (int i = 0; i < 8; i++) t += sred[i];
    sSum = t;
    if (targets) {  // CE: -(l[t] - max - log(sum)) / B ; read before overwrite
      float lt = rp[targets[row]];
      atomicAdd(accum, -(lt - mx - logf(t)) * invB);
    }
  }
  __syncthreads();
  float inv = 1.0f / sSum;
  for (int c = tid; c < C; c += 256) rp[c] = expf(rp[c] - mx) * inv;
}

// ---------------- JSD(agg(child), parent), contiguous fanout groups ----------------
__global__ __launch_bounds__(256)
void jsd_kernel(const float* __restrict__ childP, const float* __restrict__ parentP,
                int Cp, int Cc, int fan, float* __restrict__ accum, float invB) {
  size_t idx = (size_t)blockIdx.x * 256 + threadIdx.x;
  size_t total = (size_t)BATCH * Cp;
  float val = 0.0f;
  if (idx < total) {
    size_t b = idx / (size_t)Cp;
    int j = (int)(idx - b * (size_t)Cp);
    const float* ch = childP + b * (size_t)Cc + (size_t)j * fan;
    float p = 0.0f;
    for (int i = 0; i < fan; i++) p += ch[i];
    float q = parentP[b * (size_t)Cp + j];
    const float eps = 1e-8f;
    float m = 0.5f * (p + q);
    float lm = logf(m + eps);
    val = 0.5f * (p * (logf(p + eps) - lm) + q * (logf(q + eps) - lm)) * invB;
  }
  val = waveSum(val);
  __shared__ float sred[8];
  const int lane = threadIdx.x & 31, wv = threadIdx.x >> 5;
  if (lane == 0) sred[wv] = val;
  __syncthreads();
  if (threadIdx.x == 0) {
    float t = 0.0f;
    for (int i = 0; i < 8; i++) t += sred[i];
    atomicAdd(accum, t);
  }
}

// ---------------- cosine(parent_w, l2norm(segment_sum(child_w))) ----------------
__global__ __launch_bounds__(32)
void wcos_kernel(const float* __restrict__ pw, const float* __restrict__ cw,
                 int Cp, int fan, float* __restrict__ accum) {
  const int j = blockIdx.x, lane = threadIdx.x;
  float hat[16];
#pragma unroll
  for (int t = 0; t < 16; t++) hat[t] = 0.0f;
  for (int i = 0; i < fan; i++) {
    const float* cr = cw + ((size_t)j * fan + i) * FEAT;
#pragma unroll
    for (int t = 0; t < 16; t++) hat[t] += cr[lane + 32 * t];
  }
  float ss = 0.0f;
#pragma unroll
  for (int t = 0; t < 16; t++) ss += hat[t] * hat[t];
  ss = waveSum(ss);
  float inv = 1.0f / fmaxf(sqrtf(ss), 1e-12f);
  const float* pr = pw + (size_t)j * FEAT;
  float dot = 0.0f;
#pragma unroll
  for (int t = 0; t < 16; t++) dot += pr[lane + 32 * t] * hat[t];
  dot = waveSum(dot) * inv;
  if (lane == 0) atomicAdd(accum, dot / (float)Cp);
}

// =====================================================================
extern "C" void kernel_launch(void* const* d_in, const int* in_sizes, int n_in,
                              void* d_out, int out_size, void* d_ws, size_t ws_size,
                              hipStream_t stream) {
  const float* x       = (const float*)d_in[0];
  const int*   targets = (const int*)d_in[1];
  const float* W[5]  = {(const float*)d_in[2], (const float*)d_in[4],
                        (const float*)d_in[6], (const float*)d_in[8],
                        (const float*)d_in[10]};
  const float* bv[5] = {(const float*)d_in[3], (const float*)d_in[5],
                        (const float*)d_in[7], (const float*)d_in[9],
                        (const float*)d_in[11]};

  const int Cs[5]   = {20, 100, 500, 2000, 10000};
  const int Cpad[5] = {128, 128, 512, 2048, 10112};  // multiples of 128 (block N tile)

  // ---- workspace carve ----
  char* ws = (char*)d_ws;
  size_t off = 0;
  auto carve = [&](size_t bytes) -> void* {
    void* p = ws + off;
    off += (bytes + 255) & ~(size_t)255;
    return p;
  };

  float*    wn[5];
  _Float16* wh[5];
  for (int l = 0; l < 5; l++) wn[l] = (float*)   carve((size_t)Cpad[l] * FEAT * sizeof(float));
  for (int l = 0; l < 5; l++) wh[l] = (_Float16*)carve((size_t)Cpad[l] * FEAT * sizeof(_Float16));
  _Float16* xh = (_Float16*)carve((size_t)BATCH * FEAT * sizeof(_Float16));
  float* logits[5];
  for (int l = 0; l < 4; l++) logits[l] = (float*)carve((size_t)BATCH * Cs[l] * sizeof(float));
  logits[4] = (float*)d_out;  // p5 region; softmaxed in place
  float* accum = (float*)carve(16 * sizeof(float));
  float* out_loss = (float*)d_out + (size_t)BATCH * 10000;

  const float invB = 1.0f / (float)BATCH;

  // ---- pipeline ----
  zero_accum_kernel<<<1, 32, 0, stream>>>(accum);

  for (int l = 0; l < 5; l++)
    norm_rows_kernel<<<Cpad[l], 32, 0, stream>>>(W[l], wn[l], wh[l], Cs[l]);

  convert_x_kernel<<<(BATCH * FEAT) / 256, 256, 0, stream>>>(x, xh);

  for (int l = 0; l < 5; l++) {
    dim3 grid(Cpad[l] / 128, BATCH / 128);
    gemm_wmma_kernel<<<grid, 256, 0, stream>>>(xh, wh[l], bv[l], logits[l], Cs[l]);
  }

  for (int l = 0; l < 4; l++)
    softmax_kernel<<<BATCH, 256, 0, stream>>>(logits[l], Cs[l], nullptr, accum, invB);
  softmax_kernel<<<BATCH, 256, 0, stream>>>(logits[4], Cs[4], targets, &accum[0], invB);

  // jsd_45, jsd_34, jsd_23, jsd_12 -> accum[1..4]
  const int fans[4] = {5, 4, 5, 5};  // P5, P4, P3, P2
  for (int t = 0; t < 4; t++) {
    int childLvl = 4 - t, parentLvl = 3 - t;
    size_t total = (size_t)BATCH * Cs[parentLvl];
    int blocks = (int)((total + 255) / 256);
    jsd_kernel<<<blocks, 256, 0, stream>>>(logits[childLvl], logits[parentLvl],
                                           Cs[parentLvl], Cs[childLvl], fans[t],
                                           &accum[1 + t], invB);
  }

  // cosine terms -> accum[5..8]
  for (int t = 0; t < 4; t++) {
    int childLvl = 4 - t, parentLvl = 3 - t;
    wcos_kernel<<<Cs[parentLvl], 32, 0, stream>>>(wn[parentLvl], wn[childLvl],
                                                  Cs[parentLvl], fans[t],
                                                  &accum[5 + t]);
  }

  finalize_kernel<<<1, 1, 0, stream>>>(accum, out_loss);
}
